// FuySeasonTrendLayer_87943750353294
// MI455X (gfx1250) — matrix operand order repaired and checked by reading, not compiled, not run on hardware
//
#include <hip/hip_runtime.h>
#include <hip/hip_bf16.h>
#include <math.h>

// ---------------------------------------------------------------------------
// FuySeasonTrendLayer for MI455X (gfx1250, wave32, WMMA).
//   trend  = GEMM (8192x1536)x(1536x512)  [conv3 + poly folded]
//   spec   = GEMM (8192x512)x(512x512)    [DFT bins 1..255 re|im]
//   mlp    = 2x GEMM (8192x512)x(512x512) [LN -> w1 -> gelu -> w2 -> +x]
// All GEMMs: f16 A/B, f32 accumulate via v_wmma_f32_16x16x32_f16.
// ~30 GFLOP vs ~90MB traffic -> compute bound, so feed the matrix pipe and
// move tiles with the async global->LDS DMA (ASYNCcnt) when available.
// ---------------------------------------------------------------------------

typedef __attribute__((ext_vector_type(16))) _Float16     v16h;
typedef __attribute__((ext_vector_type(8)))  float        v8f;
typedef __attribute__((ext_vector_type(4)))  unsigned int u32x4;
typedef int v4i_vs __attribute__((vector_size(16)));   // matches builtin proto

union Frag16 { v16h h; u32x4 q[2]; unsigned int u[8]; };

#define B_ROWS 8192
#define L_DIM  512
#define K_TREND 1536

#if defined(__HIP_DEVICE_COMPILE__) &&                                   \
    __has_builtin(__builtin_amdgcn_global_load_async_to_lds_b128) &&     \
    __has_builtin(__builtin_amdgcn_s_wait_asynccnt)
#define USE_ASYNC_LDS 1
#else
#define USE_ASYNC_LDS 0
#endif

__device__ __forceinline__ float gelu_erf(float x) {
    return 0.5f * x * (1.0f + erff(x * 0.70710678118654752f));
}

#if USE_ASYNC_LDS
__device__ __forceinline__ void async_b128(const void* g, void* l) {
    __builtin_amdgcn_global_load_async_to_lds_b128(
        (__attribute__((address_space(1))) v4i_vs*)(uintptr_t)g,
        (__attribute__((address_space(3))) v4i_vs*)(uintptr_t)l,
        0, 0);
}
#endif

// ------------------------- weight prep kernels -----------------------------

// w1h/w2h: plain f32->f16 (already N-major: Bt[o][l] = w[o][l]).
__global__ __launch_bounds__(256) void prep_w12_kernel(
    const float* __restrict__ w1, const float* __restrict__ w2,
    _Float16* __restrict__ w1h, _Float16* __restrict__ w2h) {
    int i = blockIdx.x * 256 + threadIdx.x;           // 512*512 total
    w1h[i] = (_Float16)w1[i];
    w2h[i] = (_Float16)w2[i];
}

// Weff (N-major 512 x 1536): Weff[l][i*3+j] = poly-combination of conv3_w[l,i,k]
__global__ __launch_bounds__(256) void prep_weff_kernel(
    const float* __restrict__ c3w, _Float16* __restrict__ weff) {
    int idx = blockIdx.x * 256 + threadIdx.x;         // 512*512 (l,i) pairs
    int l = idx >> 9, i = idx & 511;
    size_t base = (size_t)l * 1536 + i * 3;
    float w0 = c3w[base + 0], w1 = c3w[base + 1], w2 = c3w[base + 2];
    weff[base + 0] = (_Float16)(0.50f * w1 + 0.25f * w0);
    weff[base + 1] = (_Float16)(0.50f * w2 + 0.25f * w1 + 0.125f * w0);
    weff[base + 2] = (_Float16)(0.25f * w2 + 0.125f * w1);
}

// DFT matrix (N-major 512 x 512): col c<255: cos(2*pi*(c+1)*t/512),
// 255<=c<510: -sin(2*pi*(c-254)*t/512), else 0.  Re X | Im X columns.
__global__ __launch_bounds__(256) void prep_dft_kernel(_Float16* __restrict__ dftB) {
    int idx = blockIdx.x * 256 + threadIdx.x;         // 512*512
    int c = idx >> 9, t = idx & 511;
    float v = 0.0f;
    const float w = 0.01227184630308513f;             // 2*pi/512
    if (c < 255) {
        int ph = ((c + 1) * t) & 511;
        v = cosf(w * (float)ph);
    } else if (c < 510) {
        int ph = ((c - 254) * t) & 511;
        v = -sinf(w * (float)ph);
    }
    dftB[(size_t)c * 512 + t] = (_Float16)v;
}

// ------------------------- per-row preprocessing ---------------------------
// For each batch row: conv1(1->3,k=3,pad1)+gelu -> Atrend (f16, packed i*3+j),
// LayerNorm -> lnh (f16), x -> xh (f16).
__global__ __launch_bounds__(256) void rowprep_kernel(
    const float* __restrict__ x,
    const float* __restrict__ c1w, const float* __restrict__ c1b,
    const float* __restrict__ lng, const float* __restrict__ lnb,
    _Float16* __restrict__ xh, _Float16* __restrict__ lnh,
    _Float16* __restrict__ atrend) {
    __shared__ float xs[514];                 // halo padded row
    __shared__ float red[256];
    __shared__ float stats[2];
    const int b = blockIdx.x, tid = threadIdx.x;
    const float* xr = x + (size_t)b * L_DIM;
    float v0 = xr[tid], v1 = xr[tid + 256];
    xs[tid + 1] = v0; xs[tid + 257] = v1;
    if (tid == 0) { xs[0] = 0.0f; xs[513] = 0.0f; }

    red[tid] = v0 + v1; __syncthreads();
    for (int s = 128; s > 0; s >>= 1) { if (tid < s) red[tid] += red[tid + s]; __syncthreads(); }
    if (tid == 0) stats[0] = red[0]; __syncthreads();
    red[tid] = v0 * v0 + v1 * v1; __syncthreads();
    for (int s = 128; s > 0; s >>= 1) { if (tid < s) red[tid] += red[tid + s]; __syncthreads(); }
    if (tid == 0) stats[1] = red[0]; __syncthreads();

    const float mu   = stats[0] * (1.0f / 512.0f);
    const float var  = stats[1] * (1.0f / 512.0f) - mu * mu;
    const float rstd = rsqrtf(var + 1e-5f);

    float wk[9], bk[3];
    #pragma unroll
    for (int j = 0; j < 9; ++j) wk[j] = c1w[j];
    #pragma unroll
    for (int j = 0; j < 3; ++j) bk[j] = c1b[j];

    #pragma unroll
    for (int h = 0; h < 2; ++h) {
        int i = tid + h * 256;
        float xm = xs[i], x0 = xs[i + 1], xp = xs[i + 2];
        #pragma unroll
        for (int j = 0; j < 3; ++j) {
            float a = fmaf(wk[j * 3 + 0], xm,
                      fmaf(wk[j * 3 + 1], x0,
                      fmaf(wk[j * 3 + 2], xp, bk[j])));
            atrend[(size_t)b * K_TREND + i * 3 + j] = (_Float16)gelu_erf(a);
        }
        float ln = (x0 - mu) * rstd * lng[i] + lnb[i];
        lnh[(size_t)b * L_DIM + i] = (_Float16)ln;
        xh [(size_t)b * L_DIM + i] = (_Float16)x0;
    }
}

// ------------------------- WMMA tiled GEMM ---------------------------------
// C(8192x512) = A(8192xK) * Bt(512xK)^T with fused epilogues.
// Workgroup: 128x128 tile, 8 waves (4x2), each wave 32x64 = 2x4 WMMA frags.
// Double-buffered LDS; tiles staged with global_load_async_to_lds_b128 when
// available (ASYNCcnt pipeline), else through-VGPR b128 copies.
// EP: 0=trend(+0.875*bias)  1=raw f32  2=gelu(+bias)->f16  3=x+acc+bias, zero-fill
template <int EP>
__global__ __launch_bounds__(256) void gemm_wmma_kernel(
    const _Float16* __restrict__ A, const _Float16* __restrict__ Bt, int K,
    float* __restrict__ outF, _Float16* __restrict__ outH,
    const float* __restrict__ bias, const float* __restrict__ xres,
    float* __restrict__ out0) {
    __shared__ alignas(16) _Float16 As[2][128][40];
    __shared__ alignas(16) _Float16 Bs[2][128][40];

    const int tid  = threadIdx.x;
    const int lane = tid & 31;
    const int wave = tid >> 5;
    const int wm = wave & 3;          // 0..3 -> M offset wm*32
    const int wn = wave >> 2;         // 0..1 -> N offset wn*64
    const int bm = blockIdx.y * 128;
    const int bn = blockIdx.x * 128;

    const int ldr = tid >> 1;                  // 0..127 tile row
    const int ldc = (tid & 1) * 16;            // 0 or 16 (16 f16 per thread)
    const _Float16* gA = A  + (size_t)(bm + ldr) * K + ldc;
    const _Float16* gB = Bt + (size_t)(bn + ldr) * K + ldc;

    v8f acc[2][4];
    #pragma unroll
    for (int i = 0; i < 2; ++i)
        #pragma unroll
        for (int j = 0; j < 4; ++j) acc[i][j] = (v8f)(0.0f);

    const int kg = lane >> 4;                  // fragment K-group
    const int ml = lane & 15;

    // stage one 128x32 A tile + 128x32 B tile into buffer `buf`
    auto stage = [&](int kt, int buf) {
#if USE_ASYNC_LDS
        async_b128(gA + kt,     &As[buf][ldr][ldc]);
        async_b128(gA + kt + 8, &As[buf][ldr][ldc] + 8);
        async_b128(gB + kt,     &Bs[buf][ldr][ldc]);
        async_b128(gB + kt + 8, &Bs[buf][ldr][ldc] + 8);
#else
        *(u32x4*)&As[buf][ldr][ldc]       = *(const u32x4*)(gA + kt);
        *(u32x4*)(&As[buf][ldr][ldc] + 8) = *(const u32x4*)(gA + kt + 8);
        *(u32x4*)&Bs[buf][ldr][ldc]       = *(const u32x4*)(gB + kt);
        *(u32x4*)(&Bs[buf][ldr][ldc] + 8) = *(const u32x4*)(gB + kt + 8);
        __builtin_prefetch(gA + kt + 32, 0, 3);
        __builtin_prefetch(gB + kt + 32, 0, 3);
#endif
    };

    const int nk = K >> 5;
    stage(0, 0);
    for (int t = 0; t < nk; ++t) {
        const int cur = t & 1;
        if (t + 1 < nk) {
            stage((t + 1) << 5, cur ^ 1);      // DMA next tile into spare buf
#if USE_ASYNC_LDS
            __builtin_amdgcn_s_wait_asynccnt(4);  // in-order: tile t landed
#endif
        } else {
#if USE_ASYNC_LDS
            __builtin_amdgcn_s_wait_asynccnt(0);
#endif
        }
        __syncthreads();

        Frag16 af[2], bf[4];
        #pragma unroll
        for (int fm = 0; fm < 2; ++fm) {       // A 16x32 frag (ISA 7.12.2 layout)
            const unsigned int* rowA = reinterpret_cast<const unsigned int*>(
                &As[cur][wm * 32 + fm * 16 + ml][0]);
            af[fm].q[0] = *(const u32x4*)(rowA + 4 * kg);
            af[fm].q[1] = *(const u32x4*)(rowA + 8 + 4 * kg);
        }
        #pragma unroll
        for (int fn = 0; fn < 4; ++fn) {       // B 32x16 frag from N-major tile
            const unsigned int* rowB = reinterpret_cast<const unsigned int*>(
                &Bs[cur][wn * 64 + fn * 16 + ml][0]);
            bf[fn].q[0] = *(const u32x4*)(rowB + 4 * kg);
            bf[fn].q[1] = *(const u32x4*)(rowB + 8 + 4 * kg);
        }
        #pragma unroll
        for (int fm = 0; fm < 2; ++fm)
            #pragma unroll
            for (int fn = 0; fn < 4; ++fn)
                acc[fm][fn] = __builtin_amdgcn_wmma_f32_16x16x32_f16(
                    false, af[fm].h, false, bf[fn].h,
                    (short)0, acc[fm][fn], false, false);
        __syncthreads();                       // buf `cur` free for t+2's DMA
    }

    // epilogue (C/D layout: lane 0-15 -> M=r, 16-31 -> M=r+8; N = lane&15)
    const int rbase = (lane >> 4) * 8;
    #pragma unroll
    for (int fm = 0; fm < 2; ++fm) {
        #pragma unroll
        for (int fn = 0; fn < 4; ++fn) {
            #pragma unroll
            for (int r = 0; r < 8; ++r) {
                const int gm = bm + wm * 32 + fm * 16 + rbase + r;
                const int gn = bn + wn * 64 + fn * 16 + ml;
                const size_t o = (size_t)gm * L_DIM + gn;
                const float v = acc[fm][fn][r];
                if constexpr (EP == 0) {            // trend: + 0.875*conv3_b
                    outF[o] = v + 0.875f * bias[gn];
                } else if constexpr (EP == 1) {     // raw DFT spectrum
                    outF[o] = v;
                } else if constexpr (EP == 2) {     // gelu(acc+b1) -> f16
                    outH[o] = (_Float16)gelu_erf(v + bias[gn]);
                } else {                            // e = x + acc + b2; seg0 = 0
                    outF[o] = xres[o] + v + bias[gn];
                    out0[o] = 0.0f;
                }
            }
        }
    }
}

// ------------------------- top-k + harmonic reconstruction -----------------
__global__ __launch_bounds__(256) void season_kernel(
    const float* __restrict__ spec, float* __restrict__ outS) {
    __shared__ float sval[256];
    __shared__ int   sidx[256];
    __shared__ float tre[5], tim[5];
    __shared__ int   tf[5];
    __shared__ int   winner;
    const int b = blockIdx.x, tid = threadIdx.x;
    const float* sp = spec + (size_t)b * L_DIM;

    float re = 0.0f, im = 0.0f, m2 = -1.0f;
    if (tid < 255) {
        re = sp[tid];
        im = sp[255 + tid];
        m2 = re * re + im * im;
    }
    bool taken = false;
    for (int it = 0; it < 5; ++it) {
        sval[tid] = (tid < 255 && !taken) ? m2 : -1.0f;
        sidx[tid] = tid;
        __syncthreads();
        for (int s = 128; s > 0; s >>= 1) {
            if (tid < s) {
                float a = sval[tid], c = sval[tid + s];
                int ia = sidx[tid], ic = sidx[tid + s];
                if (c > a || (c == a && ic < ia)) { sval[tid] = c; sidx[tid] = ic; }
            }
            __syncthreads();
        }
        if (tid == 0) winner = sidx[0];
        __syncthreads();
        if (tid == winner) { taken = true; tre[it] = re; tim[it] = im; tf[it] = tid + 1; }
        __syncthreads();
    }

    float* o = outS + (size_t)b * L_DIM;
    const float w = 0.01227184630308513f;             // 2*pi/512
    #pragma unroll
    for (int h = 0; h < 2; ++h) {
        int t = tid + h * 256;
        float s = 0.0f;
        #pragma unroll
        for (int j = 0; j < 5; ++j) {
            int ph = (tf[j] * t) & 511;
            float ang = w * (float)ph, sn, cs;
            __sincosf(ang, &sn, &cs);
            s += 2.0f * (tre[j] * cs - tim[j] * sn);
        }
        o[t] = s;
    }
}

// ---------------------------------------------------------------------------

extern "C" void kernel_launch(void* const* d_in, const int* in_sizes, int n_in,
                              void* d_out, int out_size, void* d_ws, size_t ws_size,
                              hipStream_t stream) {
    const float* x   = (const float*)d_in[0];
    const float* c1w = (const float*)d_in[1];
    const float* c1b = (const float*)d_in[2];
    const float* c3w = (const float*)d_in[3];
    const float* c3b = (const float*)d_in[4];
    const float* w1  = (const float*)d_in[5];
    const float* b1  = (const float*)d_in[6];
    const float* w2  = (const float*)d_in[7];
    const float* b2  = (const float*)d_in[8];
    const float* lng = (const float*)d_in[9];
    const float* lnb = (const float*)d_in[10];

    float* out = (float*)d_out;
    char*  ws  = (char*)d_ws;

    const size_t BL = (size_t)B_ROWS * L_DIM;         // 4,194,304
    // workspace layout (~70.3 MB total)
    size_t off = 0;
    _Float16* xh    = (_Float16*)(ws + off); off += BL * 2;
    _Float16* lnh   = (_Float16*)(ws + off); off += BL * 2;
    _Float16* m1h   = (_Float16*)(ws + off); off += BL * 2;
    _Float16* at    = (_Float16*)(ws + off); off += (size_t)B_ROWS * K_TREND * 2;
    float*    spec  = (float*)   (ws + off); off += BL * 4;
    _Float16* weffh = (_Float16*)(ws + off); off += (size_t)L_DIM * K_TREND * 2;
    _Float16* w1h   = (_Float16*)(ws + off); off += (size_t)L_DIM * L_DIM * 2;
    _Float16* w2h   = (_Float16*)(ws + off); off += (size_t)L_DIM * L_DIM * 2;
    _Float16* dftB  = (_Float16*)(ws + off); off += (size_t)L_DIM * L_DIM * 2;

    float* out_zero   = out;                          // (e - mean) == 0
    float* out_season = out + BL;
    float* out_trend  = out + 2 * BL;
    float* out_e      = out + 3 * BL;

    // weight prep (small)
    prep_w12_kernel <<<1024, 256, 0, stream>>>(w1, w2, w1h, w2h);
    prep_weff_kernel<<<1024, 256, 0, stream>>>(c3w, weffh);
    prep_dft_kernel <<<1024, 256, 0, stream>>>(dftB);

    // per-row: conv1+gelu -> Atrend, layernorm -> lnh, x -> xh
    rowprep_kernel<<<B_ROWS, 256, 0, stream>>>(x, c1w, c1b, lng, lnb, xh, lnh, at);

    const dim3 gg(L_DIM / 128, B_ROWS / 128);         // (4, 64)
    // trend GEMM (K=1536) with folded poly weights
    gemm_wmma_kernel<0><<<gg, 256, 0, stream>>>(at, weffh, K_TREND,
        out_trend, nullptr, c3b, nullptr, nullptr);
    // DFT GEMM (K=512) -> spectrum
    gemm_wmma_kernel<1><<<gg, 256, 0, stream>>>(xh, dftB, L_DIM,
        spec, nullptr, nullptr, nullptr, nullptr);
    // top-5 + harmonic reconstruction -> season
    season_kernel<<<B_ROWS, 256, 0, stream>>>(spec, out_season);
    // MLP GEMM 1: gelu(ln @ w1^T + b1) -> f16
    gemm_wmma_kernel<2><<<gg, 256, 0, stream>>>(lnh, w1h, L_DIM,
        nullptr, m1h, b1, nullptr, nullptr);
    // MLP GEMM 2: e = x + (m @ w2^T + b2); also zero-fill output[0]
    gemm_wmma_kernel<3><<<gg, 256, 0, stream>>>(m1h, w2h, L_DIM,
        out_e, nullptr, b2, x, out_zero);
}
